// LunarAttention_2362232013462
// MI455X (gfx1250) — matrix operand order repaired and checked by reading, not compiled, not run on hardware
//
#include <hip/hip_runtime.h>
#include <math.h>

// ---------------------------------------------------------------------------
// Types
// ---------------------------------------------------------------------------
typedef __bf16 bf16;
typedef __attribute__((ext_vector_type(16))) __bf16 v16bf;
typedef __attribute__((ext_vector_type(8)))  __bf16 v8bf;
typedef __attribute__((ext_vector_type(4)))  __bf16 v4bf;
typedef __attribute__((ext_vector_type(8)))  float  v8f;

union Frag {
    v16bf v;
    v8bf  h[2];
};

// ---------------------------------------------------------------------------
// WMMA helper: D = A(16x32 bf16) * B(32x16 bf16) + C(16x16 f32)
// ---------------------------------------------------------------------------
__device__ __forceinline__ v8f wmma_bf16(v16bf a, v16bf b, v8f c) {
    return __builtin_amdgcn_wmma_f32_16x16x32_bf16(
        /*neg_a=*/false, a, /*neg_b=*/false, b,
        /*c_mod=*/(short)0, c, /*reuse_a=*/false, /*reuse_b=*/false);
}

// A-fragment (16x32, M x K): lane L holds row M=(L&15);
// halfs[0..15] = K = (L>>4)*16 + 0..15 (contiguous)
__device__ __forceinline__ v16bf load_frag_a(const bf16* base, int row_stride, int lane) {
    const bf16* p = base + (size_t)(lane & 15) * row_stride + ((lane >> 4) << 4);
    Frag f;
    f.h[0] = *(const v8bf*)(p);
    f.h[1] = *(const v8bf*)(p + 8);
    return f.v;
}

// B-fragment (32x16, K x N) loaded from a [N,K] row-major matrix (i.e. B = W^T):
// lane L holds column N=(L&15); halfs[0..15] = K = (L>>4)*16 + 0..15 (contiguous)
__device__ __forceinline__ v16bf load_frag_b(const bf16* base, int row_stride, int lane) {
    const bf16* p = base + (size_t)(lane & 15) * row_stride + ((lane >> 4) << 4);
    Frag f;
    f.h[0] = *(const v8bf*)(p);
    f.h[1] = *(const v8bf*)(p + 8);
    return f.v;
}

// Async global->LDS copy of 16 bytes (ASYNCcnt-tracked), gfx1250.
__device__ __forceinline__ void async_copy_b128(unsigned lds_byte_addr, const void* gaddr) {
    asm volatile("global_load_async_to_lds_b128 %0, %1, off"
                 :: "v"(lds_byte_addr), "v"(gaddr) : "memory");
}
__device__ __forceinline__ void wait_async0() {
    asm volatile("s_wait_asynccnt 0" ::: "memory");
}
// Generic pointer to LDS byte address (low 32 bits of flat aperture address).
__device__ __forceinline__ unsigned lds_addr_of(const void* p) {
    return (unsigned)(uintptr_t)p;
}

// ---------------------------------------------------------------------------
// Kernel 1: fp32 -> bf16 conversion (vectorized x4)
// ---------------------------------------------------------------------------
__global__ __launch_bounds__(256) void cvt_f32_bf16_kernel(
    const float* __restrict__ in, bf16* __restrict__ out, int n4)
{
    int i = blockIdx.x * blockDim.x + threadIdx.x;
    if (i < n4) {
        float4 f = ((const float4*)in)[i];
        v4bf o;
        o[0] = (bf16)f.x; o[1] = (bf16)f.y; o[2] = (bf16)f.z; o[3] = (bf16)f.w;
        ((v4bf*)out)[i] = o;
    }
}

// ---------------------------------------------------------------------------
// Kernel 2: GEMM  C[M,N] = A[M,K](bf16) * W[N,K]^T(bf16) + bias
// Block: 256 threads (8 waves). Block tile 128x256, wave tile 64x64
// (16 WMMA accumulators -> 512 B loaded per WMMA).
// ---------------------------------------------------------------------------
__global__ __launch_bounds__(256) void gemm_bf16_kernel(
    const bf16* __restrict__ A, const bf16* __restrict__ W,
    const float* __restrict__ bias,
    float* __restrict__ outF, bf16* __restrict__ outB,
    int M, int N, int K)
{
    const int lane = threadIdx.x & 31;
    const int w    = threadIdx.x >> 5;
    const int wm   = w >> 2;            // 0..1
    const int wn   = w & 3;             // 0..3
    const int m0   = blockIdx.y * 128 + wm * 64;
    const int n0   = blockIdx.x * 256 + wn * 64;

    v8f c[4][4] = {};

    const bf16* Abase = A + (size_t)m0 * K;
    const bf16* Wbase = W + (size_t)n0 * K;

    for (int kc = 0; kc < K; kc += 32) {
        v16bf bq[4];
#pragma unroll
        for (int nt = 0; nt < 4; ++nt)
            bq[nt] = load_frag_b(Wbase + (size_t)(nt * 16) * K + kc, K, lane);
#pragma unroll
        for (int mt = 0; mt < 4; ++mt) {
            v16bf a = load_frag_a(Abase + (size_t)(mt * 16) * K + kc, K, lane);
#pragma unroll
            for (int nt = 0; nt < 4; ++nt)
                c[mt][nt] = wmma_bf16(a, bq[nt], c[mt][nt]);
        }
    }

    const int lh = lane >> 4;   // lane-half selects +8 rows
    const int ln = lane & 15;   // column within 16-wide tile
#pragma unroll
    for (int mt = 0; mt < 4; ++mt)
#pragma unroll
        for (int nt = 0; nt < 4; ++nt)
#pragma unroll
            for (int r = 0; r < 8; ++r) {
                int row = m0 + mt * 16 + r + lh * 8;
                int col = n0 + nt * 16 + ln;
                float v = c[mt][nt][r] + bias[col];
                if (outF) outF[(size_t)row * N + col] = v;
                else      outB[(size_t)row * N + col] = (bf16)v;
            }
}

// ---------------------------------------------------------------------------
// Kernel 3: RoPE + layout reorg.
//   mode 0: Q  -> rope, *scale, write [B,H,S,D]
//   mode 1: K  -> rope,          write [B,H,S,D]
//   mode 2: V  -> transpose to   [B,H,D,S]
// src is [B,S,H*D]. Block = 128 threads (one per d), grid = (S, H, B).
// ---------------------------------------------------------------------------
__global__ __launch_bounds__(128) void rope_reorg_kernel(
    const bf16* __restrict__ src, bf16* __restrict__ dst, int mode, float scale)
{
    const int S = 2048, H = 16, D = 128, E = 2048;
    const int d = threadIdx.x;
    const int s = blockIdx.x;
    const int h = blockIdx.y;
    const int b = blockIdx.z;

    size_t base = ((size_t)b * S + s) * E + h * D;
    float x = (float)src[base + d];

    if (mode < 2) {
        int i = d & 63;  // frequency index
        // inv_freq = 10000^(-i/64) = 2^(-i*log2(10000)/64)
        float freq = exp2f(-(float)i * (13.287712379549449f / 64.0f));
        float ang  = (float)s * freq;
        float sn, cs;
        __sincosf(ang, &sn, &cs);
        float xp = (float)src[base + (d ^ 64)];   // partner element
        float y  = (d < 64) ? (x * cs - xp * sn) : (x * cs + xp * sn);
        y *= scale;
        dst[(((size_t)b * H + h) * S + s) * D + d] = (bf16)y;
    } else {
        dst[(((size_t)b * H + h) * D + d) * S + s] = (bf16)x;
    }
}

// ---------------------------------------------------------------------------
// Kernel 4: flash-style attention with double-buffered async-LDS K/V staging.
//   Q  [B,H,S,D] bf16 (pre-scaled by 1/sqrt(D))
//   Kt [B,H,S,D] bf16
//   Vt [B,H,D,S] bf16 (transposed so P*V fragments are contiguous)
//   O  [B,S,H*D] bf16
// Block: 256 threads = 8 waves; each wave owns 16 query rows; all waves share
// one (b,h), so K/V chunks are staged into LDS once per block per k-chunk via
// global_load_async_to_lds_b128 (ASYNCcnt) with prefetch overlap.
// Grid: (S/128, H, B). No mask (reference mask is a no-op).
// ---------------------------------------------------------------------------
__global__ __launch_bounds__(256) void attn_kernel(
    const bf16* __restrict__ Q, const bf16* __restrict__ Kt,
    const bf16* __restrict__ Vt, bf16* __restrict__ O)
{
    const int S = 2048, H = 16, D = 128, E = 2048;
    // Padded rows: K rows 272B (68 dwords), V rows 80B (20 dwords) -> the
    // 16-lane fragment gathers hit 16 distinct bank groups (conflict-free b128).
    __shared__ bf16 lds_k[2][32][136];  // K chunk: 32 key rows x 128 d (+pad)
    __shared__ bf16 lds_v[2][128][40];  // V chunk: 128 d rows x 32 keys (+pad)
    __shared__ bf16 lds_p[8][16][40];   // per-wave 16x32 P tile (C->A relayout)

    const int t    = threadIdx.x;
    const int lane = t & 31;
    const int w    = t >> 5;
    const int b    = blockIdx.z;
    const int h    = blockIdx.y;
    const int q0   = blockIdx.x * 128 + w * 16;

    const size_t bh = (size_t)b * H + h;
    const bf16* Qb = Q  + (bh * S + q0) * D;
    const bf16* Kb = Kt + bh * S * D;
    const bf16* Vb = Vt + bh * D * S;

    // Cooperative staging of one 32-key chunk of K and V into LDS buffer `buf`.
    auto stage = [&](int buf, int kc) {
        // K: 32 rows x 256B = 512 x 16B chunks, 2 per thread
#pragma unroll
        for (int i = 0; i < 2; ++i) {
            int c   = t + i * 256;
            int row = c >> 4;
            int off = (c & 15) * 16;  // bytes within row
            const void* g = (const char*)(Kb + (size_t)(kc + row) * D) + off;
            async_copy_b128(lds_addr_of(&lds_k[buf][row][0]) + off, g);
        }
        // V: 128 rows x 64B = 512 x 16B chunks, 2 per thread
#pragma unroll
        for (int i = 0; i < 2; ++i) {
            int c   = t + i * 256;
            int row = c >> 2;
            int off = (c & 3) * 16;
            const void* g = (const char*)(Vb + (size_t)row * S + kc) + off;
            async_copy_b128(lds_addr_of(&lds_v[buf][row][0]) + off, g);
        }
    };

    // Q fragments: 4 chunks of K-dim (D=128), loaded once.
    v16bf qf[4];
#pragma unroll
    for (int dc = 0; dc < 4; ++dc)
        qf[dc] = load_frag_a(Qb + dc * 32, D, lane);

    v8f o[8] = {};
    float m_i[8], l_i[8];
#pragma unroll
    for (int r = 0; r < 8; ++r) { m_i[r] = -INFINITY; l_i[r] = 0.0f; }

    const int lh = lane >> 4;
    const int ln = lane & 15;

    stage(0, 0);  // prime the pipeline

    for (int kc = 0; kc < S; kc += 32) {
        const int buf = (kc >> 5) & 1;
        wait_async0();       // current chunk resident in LDS (this wave's copies)
        __syncthreads();     // ...and everyone else's
        if (kc + 32 < S) stage(buf ^ 1, kc + 32);   // prefetch next chunk

        // ---- S = Q * K^T for 32 key positions (2 N-tiles) from LDS ----
        v8f s0 = {}, s1 = {};
#pragma unroll
        for (int dc = 0; dc < 4; ++dc) {
            Frag k0, k1;
            {
                const bf16* p0 = &lds_k[buf][ln][dc * 32 + lh * 16];
                k0.h[0] = *(const v8bf*)(p0);
                k0.h[1] = *(const v8bf*)(p0 + 8);
                const bf16* p1 = &lds_k[buf][16 + ln][dc * 32 + lh * 16];
                k1.h[0] = *(const v8bf*)(p1);
                k1.h[1] = *(const v8bf*)(p1 + 8);
            }
            s0 = wmma_bf16(qf[dc], k0.v, s0);
            s1 = wmma_bf16(qf[dc], k1.v, s1);
        }

        // ---- online softmax in C-fragment layout ----
#pragma unroll
        for (int r = 0; r < 8; ++r) {
            float vmax = fmaxf(s0[r], s1[r]);
#pragma unroll
            for (int off = 8; off >= 1; off >>= 1)
                vmax = fmaxf(vmax, __shfl_xor(vmax, off, 16));
            float mnew  = fmaxf(m_i[r], vmax);
            float alpha = __expf(m_i[r] - mnew);
            float p0    = __expf(s0[r] - mnew);
            float p1    = __expf(s1[r] - mnew);
            float rs    = p0 + p1;
#pragma unroll
            for (int off = 8; off >= 1; off >>= 1)
                rs += __shfl_xor(rs, off, 16);
            l_i[r] = l_i[r] * alpha + rs;
            m_i[r] = mnew;
#pragma unroll
            for (int tt = 0; tt < 8; ++tt) o[tt][r] *= alpha;

            // stash P (bf16) in C-layout: row = r + lh*8, col = ln / 16+ln
            int mrow = r + lh * 8;
            lds_p[w][mrow][ln]      = (bf16)p0;
            lds_p[w][mrow][16 + ln] = (bf16)p1;
        }

        // Per-wave LDS ops are in-order; wait for DS stores before re-reading.
        __asm__ volatile("s_wait_dscnt 0" ::: "memory");

        // ---- reload P as an A-fragment (16x32) ----
        Frag pf;
        {
            const bf16* p = &lds_p[w][ln][lh * 16];
            pf.h[0] = *(const v8bf*)(p);
            pf.h[1] = *(const v8bf*)(p + 8);
        }

        // ---- O += P * V (V fragments from LDS) ----
#pragma unroll
        for (int tt = 0; tt < 8; ++tt) {
            Frag vf;
            const bf16* p = &lds_v[buf][tt * 16 + ln][lh * 16];
            vf.h[0] = *(const v8bf*)(p);
            vf.h[1] = *(const v8bf*)(p + 8);
            o[tt] = wmma_bf16(pf.v, vf.v, o[tt]);
        }

        __syncthreads();   // everyone done reading `buf` before it is re-staged
    }

    // ---- normalize and write O [B,S,H*D] ----
#pragma unroll
    for (int r = 0; r < 8; ++r) {
        float inv = 1.0f / l_i[r];
        int row = q0 + r + lh * 8;
#pragma unroll
        for (int tt = 0; tt < 8; ++tt)
            O[((size_t)b * S + row) * E + h * D + tt * 16 + ln] = (bf16)(o[tt][r] * inv);
    }
}

// ---------------------------------------------------------------------------
// Host orchestration
// ---------------------------------------------------------------------------
extern "C" void kernel_launch(void* const* d_in, const int* in_sizes, int n_in,
                              void* d_out, int out_size, void* d_ws, size_t ws_size,
                              hipStream_t stream) {
    const int B = 2, S = 2048, E = 2048, H = 16, D = 128;
    const int M = B * S;   // 4096
    const int N = E;       // 2048
    const int K = E;       // 2048

    const float* hs = (const float*)d_in[0];
    const float* Wq = (const float*)d_in[1];
    const float* bq = (const float*)d_in[2];
    const float* Wk = (const float*)d_in[3];
    const float* bk = (const float*)d_in[4];
    const float* Wv = (const float*)d_in[5];
    const float* bv = (const float*)d_in[6];
    const float* Wo = (const float*)d_in[7];
    const float* bo = (const float*)d_in[8];
    float* out = (float*)d_out;

    char* ws = (char*)d_ws;
    const size_t MB = 1024ull * 1024ull;
    // Workspace layout (128 MiB total, with reuse):
    bf16* Xbf = (bf16*)(ws + 0 * MB);      // 16 MB  (reused for Qr after GEMMs)
    bf16* Wqb = (bf16*)(ws + 16 * MB);     //  8 MB
    bf16* Wkb = (bf16*)(ws + 24 * MB);     //  8 MB
    bf16* Wvb = (bf16*)(ws + 32 * MB);     //  8 MB
    bf16* Wob = (bf16*)(ws + 40 * MB);     //  8 MB
    bf16* Qbf = (bf16*)(ws + 48 * MB);     // 16 MB  (reused for Obf after reorg)
    bf16* Kbf = (bf16*)(ws + 64 * MB);     // 16 MB
    bf16* Vbf = (bf16*)(ws + 80 * MB);     // 16 MB
    bf16* Kr  = (bf16*)(ws + 96 * MB);     // 16 MB  [B,H,S,D]
    bf16* Vt  = (bf16*)(ws + 112 * MB);    // 16 MB  [B,H,D,S]
    bf16* Qr  = Xbf;                       // [B,H,S,D], reuses X buffer
    bf16* Obf = Qbf;                       // [B,S,E], reuses Q buffer

    // 1) fp32 -> bf16 conversions
    {
        int n4 = (M * K) / 4;
        cvt_f32_bf16_kernel<<<n4 / 256, 256, 0, stream>>>(hs, Xbf, n4);
        int w4 = (N * K) / 4;
        cvt_f32_bf16_kernel<<<w4 / 256, 256, 0, stream>>>(Wq, Wqb, w4);
        cvt_f32_bf16_kernel<<<w4 / 256, 256, 0, stream>>>(Wk, Wkb, w4);
        cvt_f32_bf16_kernel<<<w4 / 256, 256, 0, stream>>>(Wv, Wvb, w4);
        cvt_f32_bf16_kernel<<<w4 / 256, 256, 0, stream>>>(Wo, Wob, w4);
    }

    // 2) Q/K/V projections (bf16 output)
    {
        dim3 grid(N / 256, M / 128);
        gemm_bf16_kernel<<<grid, 256, 0, stream>>>(Xbf, Wqb, bq, nullptr, Qbf, M, N, K);
        gemm_bf16_kernel<<<grid, 256, 0, stream>>>(Xbf, Wkb, bk, nullptr, Kbf, M, N, K);
        gemm_bf16_kernel<<<grid, 256, 0, stream>>>(Xbf, Wvb, bv, nullptr, Vbf, M, N, K);
    }

    // 3) RoPE + reorg (Q scaled by 1/sqrt(D))
    {
        dim3 grid(S, H, B);
        const float qscale = 0.08838834764831845f;  // 1/sqrt(128)
        rope_reorg_kernel<<<grid, 128, 0, stream>>>(Qbf, Qr, 0, qscale);
        rope_reorg_kernel<<<grid, 128, 0, stream>>>(Kbf, Kr, 1, 1.0f);
        rope_reorg_kernel<<<grid, 128, 0, stream>>>(Vbf, Vt, 2, 1.0f);
    }

    // 4) attention (flash-style, async-LDS staged K/V, full softmax)
    {
        dim3 grid(S / 128, H, B);
        attn_kernel<<<grid, 256, 0, stream>>>(Qr, Kr, Vt, Obf);
    }

    // 5) output projection -> fp32 d_out
    {
        dim3 grid(N / 256, M / 128);
        gemm_bf16_kernel<<<grid, 256, 0, stream>>>(Obf, Wob, bo, out, nullptr, M, N, K);
    }
}
